// SHMCModule_55894704390149
// MI455X (gfx1250) — matrix operand order repaired and checked by reading, not compiled, not run on hardware
//
#include <hip/hip_runtime.h>
#include <stdint.h>

// Problem constants (from reference)
#define B_   16384
#define D_   4096
#define d_   512
#define L_   16384
#define KKEEP 153

typedef __attribute__((ext_vector_type(16))) __bf16 v16bf;
typedef __attribute__((ext_vector_type(8)))  float  v8f;

union BV { uint4 u[2]; v16bf v; };

// ---- gfx1250 async global->LDS path (guarded; fallback = plain copy) ----
#if defined(__has_builtin)
# if __has_builtin(__builtin_amdgcn_global_load_async_to_lds_b128)
#  define HAVE_ASYNC 1
# endif
#endif
#ifndef HAVE_ASYNC
# define HAVE_ASYNC 0
#endif

#if HAVE_ASYNC
// Builtin signature (from compiler diagnostic): first param is
// 'int __attribute__((vector_size(16))) __device__ *' (AS1), second is the
// LDS-side pointer (AS3). Cast via uintptr_t to exactly-matching types.
typedef int v4i_async __attribute__((vector_size(16)));
typedef __attribute__((address_space(1))) v4i_async* g_v4i_ptr;
typedef __attribute__((address_space(3))) v4i_async* l_v4i_ptr;
__device__ __forceinline__ void async_cp16(const void* g, void* l) {
  __builtin_amdgcn_global_load_async_to_lds_b128(
      (g_v4i_ptr)(uintptr_t)g, (l_v4i_ptr)(uintptr_t)l, 0, 0);
}
# if __has_builtin(__builtin_amdgcn_s_wait_asynccnt)
#  define WAIT_ASYNC(n) __builtin_amdgcn_s_wait_asynccnt(n)
# else
#  define WAIT_ASYNC(n) asm volatile("s_wait_asynccnt %0" :: "n"(n))
# endif
#endif

__device__ __forceinline__ uint32_t f2bf_pack(float lo, float hi) {
  // round-to-nearest-even fp32 -> bf16, packed pair
  uint32_t a = __float_as_uint(lo);
  uint32_t b = __float_as_uint(hi);
  a = (a + 0x7FFFu + ((a >> 16) & 1u)) >> 16;
  b = (b + 0x7FFFu + ((b >> 16) & 1u)) >> 16;
  return (a & 0xFFFFu) | (b << 16);
}
__device__ __forceinline__ uint16_t f2bf(float f) {
  uint32_t u = __float_as_uint(f);
  u = (u + 0x7FFFu + ((u >> 16) & 1u)) >> 16;
  return (uint16_t)u;
}
__device__ __forceinline__ float bf2f(uint32_t h) {
  return __uint_as_float(h << 16);
}

// ---------------- prep: W_proj fp32 -> bf16 ----------------
__global__ void k_prep_w(const float* __restrict__ W, uint32_t* __restrict__ Wb) {
  int i = blockIdx.x * blockDim.x + threadIdx.x;
  int base = i * 4;
  if (base < d_ * D_) {
    float4 f = *(const float4*)(W + base);
    uint2 o;
    o.x = f2bf_pack(f.x, f.y);
    o.y = f2bf_pack(f.z, f.w);
    *(uint2*)(Wb + base / 2) = o;
  }
}

// ---------------- prep: landmarks fp32 -> bf16 + ||l||^2 ----------------
__global__ void __launch_bounds__(128) k_prep_lm(const float* __restrict__ lm,
                                                 uint32_t* __restrict__ lmb,
                                                 float* __restrict__ lnorm) {
  int row = blockIdx.x;          // one landmark row per block
  int t   = threadIdx.x;         // 128 threads, 4 floats each
  const float* src = lm + (size_t)row * d_;
  float4 f = *(const float4*)(src + t * 4);
  uint2 o;
  o.x = f2bf_pack(f.x, f.y);
  o.y = f2bf_pack(f.z, f.w);
  *(uint2*)(lmb + (size_t)row * (d_ / 2) + t * 2) = o;
  float s = f.x * f.x + f.y * f.y + f.z * f.z + f.w * f.w;
  for (int m = 1; m < 32; m <<= 1) s += __shfl_xor(s, m, 32);
  __shared__ float wsum[4];
  if ((t & 31) == 0) wsum[t >> 5] = s;
  __syncthreads();
  if (t == 0) lnorm[row] = wsum[0] + wsum[1] + wsum[2] + wsum[3];
}

// ---------------- GEMM1: compressed = x @ W^T (bf16 WMMA, f32 accum) ----------------
// grid = B/16 blocks, 256 threads (8 waves). LDS: 16x4096 bf16 A-tile (128 KB).
// K-outer loop: one A ds_load pair feeds 4 WMMAs (4 N-tiles per wave).
__global__ void __launch_bounds__(256) k_gemm1(const float* __restrict__ x,
                                               const uint32_t* __restrict__ Wb,
                                               uint16_t* __restrict__ comp) {
  extern __shared__ uint8_t smem[];
  uint32_t* As = (uint32_t*)smem;   // 16 rows x 2048 uints (4096 bf16)
  int tid = threadIdx.x;
  int m0  = blockIdx.x * 16;

  // stage x rows -> bf16 in LDS (16*4096 elems = 16384 uint2-chunks)
  const float* xrow = x + (size_t)m0 * D_;
  for (int it = 0; it < 64; ++it) {
    int flat = it * 256 + tid;                 // float4 / uint2 chunk index
    float4 f = *(const float4*)(xrow + (size_t)flat * 4);
    uint2 o;
    o.x = f2bf_pack(f.x, f.y);
    o.y = f2bf_pack(f.z, f.w);
    *(uint2*)(As + flat * 2) = o;
  }
  __syncthreads();

  int w = tid >> 5, lane = tid & 31, col = lane & 15, kh = lane >> 4;
  const uint32_t* arow  = As + (size_t)col * (D_ / 2);
  const uint32_t* wbase = Wb + (size_t)(w * 16 + col) * (D_ / 2);

  v8f acc[4] = {};
  for (int ks = 0; ks < 128; ++ks) {
    BV a;
    // A (16x32 bf16): lane holds row=col; K chunks split by lane-half
    a.u[0] = *(const uint4*)(arow + ks * 16 + kh * 4);
    a.u[1] = *(const uint4*)(arow + ks * 16 + 8 + kh * 4);
#pragma unroll
    for (int p = 0; p < 4; ++p) {
      // B (32x16 bf16): lane holds column = W row; K chunks split by lane-half
      const uint32_t* wp = wbase + (size_t)p * 128 * (D_ / 2) + ks * 16 + kh * 4;
      BV b;
      b.u[0] = *(const uint4*)(wp);
      b.u[1] = *(const uint4*)(wp + 8);
      acc[p] = __builtin_amdgcn_wmma_f32_16x16x32_bf16(false, a.v, false, b.v,
                                                       (short)0, acc[p], false, false);
    }
  }
#pragma unroll
  for (int p = 0; p < 4; ++p) {
    int n0 = p * 128 + w * 16;
    // C layout: N = lane%16, M = (lane/16)*8 + r
    uint16_t* dst = comp + (size_t)(m0 + kh * 8) * d_ + n0 + col;
#pragma unroll
    for (int r = 0; r < 8; ++r) dst[(size_t)r * d_] = f2bf(acc[p][r]);
  }
}

// ---------------- column |.| partial sums (deterministic, coalesced) ----------------
__global__ void __launch_bounds__(256) k_colabs_part(const uint16_t* __restrict__ comp,
                                                     float* __restrict__ partial) {
  int blk = blockIdx.x;   // 128 blocks x 128 rows each
  int t   = threadIdx.x;  // thread t owns columns 2t, 2t+1
  const uint32_t* base = (const uint32_t*)comp + (size_t)blk * 128 * (d_ / 2) + t;
  float s0 = 0.f, s1 = 0.f;
  for (int r = 0; r < 128; ++r) {
    uint32_t v = base[(size_t)r * (d_ / 2)];
    s0 += fabsf(bf2f(v & 0xFFFFu));
    s1 += fabsf(bf2f(v >> 16));
  }
  partial[(size_t)blk * d_ + 2 * t]     = s0;
  partial[(size_t)blk * d_ + 2 * t + 1] = s1;
}

// ---------------- top-k via rank, emit packed 16-bit AND-mask ----------------
__global__ void __launch_bounds__(512) k_topk(const float* __restrict__ partial,
                                              const float* __restrict__ mw,
                                              uint32_t* __restrict__ mask_packed) {
  __shared__ float imp[512];
  __shared__ uint16_t keep[512];
  int j = threadIdx.x;
  float s = 0.f;
  for (int i = 0; i < 128; ++i) s += partial[(size_t)i * d_ + j];
  imp[j] = s * (1.0f / (float)B_) + mw[j];
  __syncthreads();
  float me = imp[j];
  int cnt = 0;
  for (int i = 0; i < 512; ++i) {
    float o = imp[i];
    cnt += (o > me) || (o == me && i < j);
  }
  keep[j] = (cnt < KKEEP) ? 0xFFFFu : 0u;
  __syncthreads();
  if (j < 256)
    mask_packed[j] = (uint32_t)keep[2 * j] | ((uint32_t)keep[2 * j + 1] << 16);
}

// ---------------- GEMM2 + fused argmin + gather ----------------
// grid = B/128 blocks, 256 threads (8 waves).
// LDS: A 128x512 bf16 (128 KB) + double-buffered B stage 2 x 64x512 bf16
// (2 x 64 KB, filled by GLOBAL_LOAD_ASYNC_TO_LDS_B128) + nearest[128].
__global__ void __launch_bounds__(256) k_gemm2_argmin(
    const uint16_t* __restrict__ comp, const uint32_t* __restrict__ mask_packed,
    const uint32_t* __restrict__ lmb, const float* __restrict__ lnorm,
    const float* __restrict__ lm_f32, float* __restrict__ out) {
  extern __shared__ uint8_t smem[];
  uint32_t* As = (uint32_t*)smem;                // 128 x 256 uints
  uint32_t* Bs = As + 128 * 256;                 // 2 x 64 x 256 uints (ping-pong)
  int* nearest = (int*)(Bs + 2 * 64 * 256);      // 128 ints

  int tid = threadIdx.x;
  int r0  = blockIdx.x * 128;

  // stage masked A (sparse = compressed & mask) into LDS
  const uint32_t* compu = (const uint32_t*)comp + (size_t)r0 * (d_ / 2);
  for (int it = 0; it < 128; ++it) {
    int flat = it * 256 + tid;
    As[flat] = compu[flat] & mask_packed[flat & 255];
  }

  int w = tid >> 5, lane = tid & 31, col = lane & 15, kh = lane >> 4;
  const uint32_t* arow = As + (size_t)(w * 16 + col) * (d_ / 2);

  float best[8];
  int   bidx[8];
#pragma unroll
  for (int r = 0; r < 8; ++r) { best[r] = 3.4e38f; bidx[r] = 0; }

  const int NG = L_ / 64;   // 256 groups of 64 landmarks

#if HAVE_ASYNC
  { // prologue: async-stage group 0 into buffer 0 (16 b128 per wave)
    const uint4* src = (const uint4*)lmb;
    uint4* dst = (uint4*)Bs;
    for (int it = 0; it < 16; ++it) {
      int chunk = it * 256 + tid;
      async_cp16(src + chunk, dst + chunk);
    }
  }
#endif

  for (int g = 0; g < NG; ++g) {
    int cur = g & 1;
    const uint32_t* bbase = Bs + (size_t)cur * 64 * 256;

#if HAVE_ASYNC
    if (g + 1 < NG) {
      // issue next group into the other buffer (safe: end-of-iter barrier of
      // iteration g-1 guarantees all readers of that buffer are done)
      const uint4* src = (const uint4*)(lmb + (size_t)(g + 1) * 64 * (d_ / 2));
      uint4* dst = (uint4*)(Bs + (size_t)(1 - cur) * 64 * 256);
      for (int it = 0; it < 16; ++it) {
        int chunk = it * 256 + tid;
        async_cp16(src + chunk, dst + chunk);
      }
      WAIT_ASYNC(16);   // retire current group's (older, in-order) transfers
    } else {
      WAIT_ASYNC(0);
    }
    __syncthreads();
#else
    __syncthreads();
    {
      const uint4* src = (const uint4*)(lmb + (size_t)g * 64 * (d_ / 2));
      uint4* dst = (uint4*)bbase;
      for (int it = 0; it < 16; ++it) {
        int flat = it * 256 + tid;
        dst[flat] = src[flat];
      }
    }
    __syncthreads();
#endif

    // 4 N-subtiles (64 columns), K-outer: one A load feeds 4 WMMAs
    v8f acc[4] = {};
    const uint32_t* brow = bbase + (size_t)col * (d_ / 2);
    for (int ks = 0; ks < 16; ++ks) {
      BV a;
      a.u[0] = *(const uint4*)(arow + ks * 16 + kh * 4);
      a.u[1] = *(const uint4*)(arow + ks * 16 + 8 + kh * 4);
#pragma unroll
      for (int s = 0; s < 4; ++s) {
        const uint32_t* bp = brow + (size_t)s * 16 * (d_ / 2) + ks * 16 + kh * 4;
        BV b;
        b.u[0] = *(const uint4*)(bp);
        b.u[1] = *(const uint4*)(bp + 8);
        acc[s] = __builtin_amdgcn_wmma_f32_16x16x32_bf16(false, a.v, false, b.v,
                                                         (short)0, acc[s], false, false);
      }
    }
#pragma unroll
    for (int s = 0; s < 4; ++s) {
      int li = g * 64 + s * 16 + col;           // landmark index for this lane
      float ln = lnorm[li];
#pragma unroll
      for (int r = 0; r < 8; ++r) {
        float dd = ln - 2.0f * acc[s][r];       // ||l||^2 - 2 s.l (argmin-equivalent)
        if (dd < best[r] || (dd == best[r] && li < bidx[r])) {
          best[r] = dd; bidx[r] = li;
        }
      }
    }
    __syncthreads();   // all reads of bbase done before it is refilled
  }

  // min-reduce across the 16 lanes that share the same M rows
  for (int m = 1; m < 16; m <<= 1) {
#pragma unroll
    for (int r = 0; r < 8; ++r) {
      float ob = __shfl_xor(best[r], m, 32);
      int   oi = __shfl_xor(bidx[r], m, 32);
      if (ob < best[r] || (ob == best[r] && oi < bidx[r])) {
        best[r] = ob; bidx[r] = oi;
      }
    }
  }
  if (col == 0) {
    int rowb = w * 16 + kh * 8;   // lane0 -> rows +0..7, lane16 -> rows +8..15
#pragma unroll
    for (int r = 0; r < 8; ++r) nearest[rowb + r] = bidx[r];
  }
  __syncthreads();

  // gather winning landmark rows (fp32) to output, coalesced float4
  for (int it = 0; it < 64; ++it) {
    int flat4 = it * 256 + tid;        // 128 rows * 128 float4 per row
    int row = flat4 >> 7, c4 = flat4 & 127;
    float4 v = *(const float4*)(lm_f32 + (size_t)nearest[row] * d_ + c4 * 4);
    *(float4*)(out + (size_t)(r0 + row) * d_ + c4 * 4) = v;
  }
}

extern "C" void kernel_launch(void* const* d_in, const int* in_sizes, int n_in,
                              void* d_out, int out_size, void* d_ws, size_t ws_size,
                              hipStream_t stream) {
  (void)in_sizes; (void)n_in; (void)out_size; (void)ws_size;
  const float* x  = (const float*)d_in[0];   // [B, D]
  const float* W  = (const float*)d_in[1];   // [d, D]
  const float* mw = (const float*)d_in[2];   // [d]
  const float* lm = (const float*)d_in[3];   // [L, d]
  float* out = (float*)d_out;                // [B, d]

  uint8_t* ws = (uint8_t*)d_ws;
  size_t off = 0;
  uint32_t* Wb      = (uint32_t*)(ws + off); off += (size_t)d_ * D_ * 2;   //  4 MB
  uint32_t* lmb     = (uint32_t*)(ws + off); off += (size_t)L_ * d_ * 2;   // 16 MB
  float*    lnorm   = (float*)   (ws + off); off += (size_t)L_ * 4;        // 64 KB
  uint16_t* comp    = (uint16_t*)(ws + off); off += (size_t)B_ * d_ * 2;   // 16 MB
  float*    partial = (float*)   (ws + off); off += (size_t)128 * d_ * 4;  // 256 KB
  uint32_t* maskp   = (uint32_t*)(ws + off); off += 256 * 4;               //  1 KB

  k_prep_w<<<(d_ * D_ / 4 + 255) / 256, 256, 0, stream>>>(W, Wb);
  k_prep_lm<<<L_, 128, 0, stream>>>(lm, lmb, lnorm);
  k_gemm1<<<B_ / 16, 256, (size_t)16 * D_ * 2, stream>>>(x, Wb, comp);
  k_colabs_part<<<128, 256, 0, stream>>>(comp, partial);
  k_topk<<<1, 512, 0, stream>>>(partial, mw, maskp);
  size_t smem2 = (size_t)128 * 256 * 4 * 2 + 128 * 4;
  k_gemm2_argmin<<<B_ / 128, 256, smem2, stream>>>(comp, maskp, lmb, lnorm, lm, out);
}
